// Gemma2Attention_70480413327841
// MI455X (gfx1250) — matrix operand order repaired and checked by reading, not compiled, not run on hardware
//
#include <hip/hip_runtime.h>

typedef __attribute__((ext_vector_type(16))) __bf16 v16bf;
typedef __attribute__((ext_vector_type(8)))  float  v8f;
typedef __attribute__((ext_vector_type(4)))  int    v4i;

#define S_LEN   4096
#define HDIM    3584
#define NQH     16
#define NKVH    8
#define DHEAD   256
#define WINDOW  2048
#define QKV_N   8192      // NQ*D + 2*NKV*D
#define CTX_N   4096      // NQ*D
#define SOFTCAP 50.0f
#define SCALEF  0.0625f   // 256^-0.5

// CDNA5 async global->LDS path (ASYNCcnt), with safe fallback.
#if defined(__gfx1250__) && \
    __has_builtin(__builtin_amdgcn_global_load_async_to_lds_b128) && \
    __has_builtin(__builtin_amdgcn_s_wait_asynccnt)
#define HAS_ASYNC_LDS 1
#else
#define HAS_ASYNC_LDS 0
#endif

#if HAS_ASYNC_LDS
typedef __attribute__((address_space(1))) v4i* gv4p_t;   // global int4*
typedef __attribute__((address_space(3))) v4i* lv4p_t;   // LDS int4*
__device__ __forceinline__ void async_cp_b128(const void* g, void* l) {
  __builtin_amdgcn_global_load_async_to_lds_b128(
      (gv4p_t)(void*)g, (lv4p_t)l, 0, 0);
}
__device__ __forceinline__ void async_wait0() {
  __builtin_amdgcn_s_wait_asynccnt(0);
}
#else
__device__ __forceinline__ void async_cp_b128(const void* g, void* l) {
  *(uint4*)l = *(const uint4*)g;
}
__device__ __forceinline__ void async_wait0() {}
#endif

__device__ __forceinline__ unsigned short f2bf(float f) {
  unsigned int u = __float_as_uint(f);
  u += 0x7fffu + ((u >> 16) & 1u);           // round-to-nearest-even
  return (unsigned short)(u >> 16);
}
__device__ __forceinline__ float bf2f(unsigned short h) {
  return __uint_as_float(((unsigned int)h) << 16);
}

union Frag16 { uint4 q[2]; v16bf v; };       // 16 bf16 operand fragment (32B)

// ---------------------------------------------------------------------------
// Generic bf16 WMMA GEMM: C[M,N] = A[M,K] * B[K,N]
// B is up to 3 concatenated f32 matrices split at column nsplit1/nsplit2.
// ---------------------------------------------------------------------------
template<bool A_BF16, bool OUT_BF16>
__global__ __launch_bounds__(256)
void gemm_kernel(const void* __restrict__ Aptr,
                 const float* __restrict__ B0, int pitch0,
                 const float* __restrict__ B1, int pitch1,
                 const float* __restrict__ B2, int pitch2,
                 int nsplit1, int nsplit2,
                 void* __restrict__ Cptr,
                 int M, int N, int K)
{
  constexpr int BM = 128, BN = 128, BK = 32;
  __shared__ unsigned short lA[BM * BK];     // [row][k]  8 KB
  __shared__ unsigned short lB[BK * BN];     // [k][col]  8 KB

  const int t    = threadIdx.x;
  const int m0   = blockIdx.y * BM;
  const int n0   = blockIdx.x * BN;
  const int lane = t & 31;
  const int wid  = t >> 5;
  const int wm   = wid >> 2;                 // 0..1 (64 rows each)
  const int wn   = wid & 3;                  // 0..3 (32 cols each)
  const int half = lane >> 4;
  const int l16  = lane & 15;

  v8f acc[4][2];
  #pragma unroll
  for (int mi = 0; mi < 4; mi++)
    #pragma unroll
    for (int ni = 0; ni < 2; ni++) acc[mi][ni] = {};

  for (int k0 = 0; k0 < K; k0 += BK) {
    // ---- stage A tile (128x32) : each thread 16 elements ----
    {
      const int row = t >> 1;
      const int cs  = (t & 1) * 16;
      if (A_BF16) {
        const unsigned short* A = (const unsigned short*)Aptr;
        const unsigned short* g = A + (size_t)(m0 + row) * K + k0 + cs;
        async_cp_b128(g,     &lA[row * BK + cs]);       // ASYNCcnt copy
        async_cp_b128(g + 8, &lA[row * BK + cs + 8]);
        if (k0 + BK < K)
          __builtin_prefetch(g + BK, 0, 3);
      } else {
        const float* A = (const float*)Aptr;
        const float4* g = (const float4*)(A + (size_t)(m0 + row) * K + k0 + cs);
        #pragma unroll
        for (int i = 0; i < 4; i++) {
          float4 f = g[i];
          lA[row * BK + cs + i * 4 + 0] = f2bf(f.x);
          lA[row * BK + cs + i * 4 + 1] = f2bf(f.y);
          lA[row * BK + cs + i * 4 + 2] = f2bf(f.z);
          lA[row * BK + cs + i * 4 + 3] = f2bf(f.w);
        }
        if (k0 + BK < K)
          __builtin_prefetch(A + (size_t)(m0 + row) * K + k0 + BK + cs, 0, 3);
      }
    }
    // ---- stage B tile (32x128) f32 -> bf16 : each thread 16 elements ----
    {
      const int row = t >> 3;
      const int cs  = (t & 7) * 16;
      const int gn  = n0 + cs;
      const float* bp; int pitch, col;
      if (gn < nsplit1)      { bp = B0; pitch = pitch0; col = gn; }
      else if (gn < nsplit2) { bp = B1; pitch = pitch1; col = gn - nsplit1; }
      else                   { bp = B2; pitch = pitch2; col = gn - nsplit2; }
      const float4* g = (const float4*)(bp + (size_t)(k0 + row) * pitch + col);
      #pragma unroll
      for (int i = 0; i < 4; i++) {
        float4 f = g[i];
        lB[row * BN + cs + i * 4 + 0] = f2bf(f.x);
        lB[row * BN + cs + i * 4 + 1] = f2bf(f.y);
        lB[row * BN + cs + i * 4 + 2] = f2bf(f.z);
        lB[row * BN + cs + i * 4 + 3] = f2bf(f.w);
      }
    }
    if (A_BF16) async_wait0();               // drain ASYNCcnt before barrier
    __syncthreads();

    // ---- WMMA: 4 m-tiles x 2 n-tiles per wave ----
    #pragma unroll
    for (int ni = 0; ni < 2; ni++) {
      Frag16 b;                               // B col n=l16, K chunks per half
      const int coff = wn * 32 + ni * 16;
      #pragma unroll
      for (int s = 0; s < 16; s++)
        ((unsigned short*)&b)[s] = lB[((s & 7) + half * 8 + (s >> 3) * 16) * BN + coff + l16];
      #pragma unroll
      for (int mi = 0; mi < 4; mi++) {
        Frag16 a;                             // A row m=l16, K chunks per half
        const int rb = (wm * 64 + mi * 16 + l16) * BK;
        a.q[0] = *(const uint4*)&lA[rb + (half ? 8  : 0)];
        a.q[1] = *(const uint4*)&lA[rb + (half ? 24 : 16)];
        acc[mi][ni] = __builtin_amdgcn_wmma_f32_16x16x32_bf16(
            false, a.v, false, b.v, (short)0, acc[mi][ni], false, false);
      }
    }
    __syncthreads();
  }

  // ---- epilogue: C layout row = r + 8*half, col = l16 ----
  #pragma unroll
  for (int mi = 0; mi < 4; mi++)
    #pragma unroll
    for (int ni = 0; ni < 2; ni++)
      #pragma unroll
      for (int r = 0; r < 8; r++) {
        const int m = m0 + wm * 64 + mi * 16 + r + half * 8;
        const int n = n0 + wn * 32 + ni * 16 + l16;
        const float v = acc[mi][ni][r];
        if (OUT_BF16) ((unsigned short*)Cptr)[(size_t)m * N + n] = f2bf(v);
        else          ((float*)Cptr)[(size_t)m * N + n]          = v;
      }
}

// ---------------------------------------------------------------------------
// RoPE in place on Q (cols 0..4095) and K (cols 4096..6143) of qkv
// ---------------------------------------------------------------------------
__global__ __launch_bounds__(256)
void rope_kernel(unsigned short* __restrict__ qkv, const int* __restrict__ pos_ids)
{
  const int idx = blockIdx.x * 256 + threadIdx.x;   // s * 24 heads * 128 pairs
  const int i   = idx & 127;
  const int tmp = idx >> 7;
  const int rh  = tmp % 24;
  const int s   = tmp / 24;
  if (s >= S_LEN) return;
  const int base = (rh < NQH) ? rh * DHEAD : CTX_N + (rh - NQH) * DHEAD;
  const size_t off = (size_t)s * QKV_N + base + i;
  const float x1 = bf2f(qkv[off]);
  const float x2 = bf2f(qkv[off + 128]);
  const float pos = (float)pos_ids[s];
  const float inv = __powf(10000.0f, -((float)i) * (1.0f / 128.0f));
  const float fr  = pos * inv;
  const float c = cosf(fr), sn = sinf(fr);
  qkv[off]       = f2bf(x1 * c - x2 * sn);
  qkv[off + 128] = f2bf(x2 * c + x1 * sn);
}

// ---------------------------------------------------------------------------
// Sliding-window GQA flash attention with tanh softcap.
// Block = 4 waves x 16 query rows; one (head, 64-query) tile per block.
// ---------------------------------------------------------------------------
__global__ __launch_bounds__(128)
void attn_kernel(const unsigned short* __restrict__ qkv,
                 unsigned short* __restrict__ ctx)
{
  __shared__ unsigned short sKT[DHEAD * 32];  // K^T tile [d][key]  16 KB
  __shared__ unsigned short sV [32 * DHEAD];  // V tile  [key][d]   16 KB
  __shared__ unsigned short sP [4 * 16 * 32]; // per-wave P staging  4 KB

  const int t    = threadIdx.x;
  const int lane = t & 31;
  const int wid  = t >> 5;
  const int half = lane >> 4;
  const int l16  = lane & 15;
  const int qblk = blockIdx.x * 64;
  const int head = blockIdx.y;
  const int kvh  = head >> 1;                 // N_REP = 2
  const int q0w  = qblk + wid * 16;

  // preload Q A-fragments over the 256-wide head dim (8 chunks of K=32)
  Frag16 qf[8];
  {
    const unsigned short* qrow =
        qkv + (size_t)(q0w + l16) * QKV_N + head * DHEAD;
    #pragma unroll
    for (int ds = 0; ds < 8; ds++) {
      qf[ds].q[0] = *(const uint4*)(qrow + ds * 32 + (half ? 8  : 0));
      qf[ds].q[1] = *(const uint4*)(qrow + ds * 32 + (half ? 24 : 16));
    }
  }

  v8f o[16];
  #pragma unroll
  for (int dt = 0; dt < 16; dt++) o[dt] = {};
  float mrow[8], lrow[8];
  #pragma unroll
  for (int r = 0; r < 8; r++) { mrow[r] = -1e30f; lrow[r] = 0.0f; }

  int kstart = qblk - (WINDOW - 1);
  if (kstart < 0) kstart = 0;
  kstart &= ~31;
  const int kend = qblk + 64;

  for (int key0 = kstart; key0 < kend; key0 += 32) {
    // ---- cooperative V stage (row-major, async global->LDS) ----
    {
      const int key = t >> 2;
      const int d0  = (t & 3) * 64;
      const unsigned short* vrow =
          qkv + (size_t)(key0 + key) * QKV_N + 6144 + kvh * DHEAD + d0;
      #pragma unroll
      for (int c = 0; c < 8; c++)
        async_cp_b128(vrow + c * 8, &sV[key * DHEAD + d0 + c * 8]);
    }
    // ---- cooperative K^T stage (transpose via LDS) ----
    {
      const int key = t & 31;
      const int d0  = (t >> 5) * 64;
      const unsigned short* krow =
          qkv + (size_t)(key0 + key) * QKV_N + CTX_N + kvh * DHEAD + d0;
      #pragma unroll
      for (int c = 0; c < 8; c++) {
        union { uint4 q; unsigned short u[8]; } x;
        x.q = *(const uint4*)(krow + c * 8);
        #pragma unroll
        for (int j = 0; j < 8; j++) sKT[(d0 + c * 8 + j) * 32 + key] = x.u[j];
      }
    }
    async_wait0();                            // drain ASYNCcnt before barrier
    __syncthreads();

    // ---- S = Q K^T : two 16x16 score tiles, K-dim = 256 in 8 WMMAs each ----
    v8f sc[2];
    #pragma unroll
    for (int nt = 0; nt < 2; nt++) {
      v8f c = {};
      #pragma unroll
      for (int ds = 0; ds < 8; ds++) {
        Frag16 b;                             // B col n=key l16, K chunks per half
        #pragma unroll
        for (int s = 0; s < 16; s++)
          ((unsigned short*)&b)[s] = sKT[(ds * 32 + (s & 7) + half * 8 + (s >> 3) * 16) * 32 + nt * 16 + l16];
        c = __builtin_amdgcn_wmma_f32_16x16x32_bf16(
            false, qf[ds].v, false, b.v, (short)0, c, false, false);
      }
      sc[nt] = c;
    }

    // ---- softcap + window mask + online softmax ----
    #pragma unroll
    for (int r = 0; r < 8; r++) {
      const int qrow = q0w + r + half * 8;
      float s0 = SOFTCAP * tanhf(sc[0][r] * (SCALEF / SOFTCAP));
      float s1 = SOFTCAP * tanhf(sc[1][r] * (SCALEF / SOFTCAP));
      const int k0i = key0 + l16;
      const int k1i = key0 + 16 + l16;
      if (!((k0i <= qrow) && (k0i > qrow - WINDOW))) s0 = -1e30f;
      if (!((k1i <= qrow) && (k1i > qrow - WINDOW))) s1 = -1e30f;
      float tm = fmaxf(s0, s1);
      tm = fmaxf(tm, __shfl_xor(tm, 1, 32));
      tm = fmaxf(tm, __shfl_xor(tm, 2, 32));
      tm = fmaxf(tm, __shfl_xor(tm, 4, 32));
      tm = fmaxf(tm, __shfl_xor(tm, 8, 32));
      const float mnew  = fmaxf(mrow[r], tm);
      const float alpha = __expf(mrow[r] - mnew);
      const float p0 = __expf(s0 - mnew);
      const float p1 = __expf(s1 - mnew);
      float ps = p0 + p1;
      ps += __shfl_xor(ps, 1, 32);
      ps += __shfl_xor(ps, 2, 32);
      ps += __shfl_xor(ps, 4, 32);
      ps += __shfl_xor(ps, 8, 32);
      lrow[r] = lrow[r] * alpha + ps;
      mrow[r] = mnew;
      #pragma unroll
      for (int dt = 0; dt < 16; dt++) o[dt][r] *= alpha;
      const int prow = r + half * 8;
      sP[wid * 512 + prow * 32 + l16]      = f2bf(p0);
      sP[wid * 512 + prow * 32 + 16 + l16] = f2bf(p1);
    }

    // ---- O += P V : re-read P in A layout (wave-private LDS, DS in-order) ----
    Frag16 pa;
    {
      const unsigned short* pb = &sP[wid * 512 + l16 * 32];
      pa.q[0] = *(const uint4*)(pb + (half ? 8  : 0));
      pa.q[1] = *(const uint4*)(pb + (half ? 24 : 16));
    }
    #pragma unroll
    for (int dt = 0; dt < 16; dt++) {
      Frag16 b;                               // B col d=l16, key chunks per half
      #pragma unroll
      for (int s = 0; s < 16; s++)
        ((unsigned short*)&b)[s] = sV[((s & 7) + half * 8 + (s >> 3) * 16) * DHEAD + dt * 16 + l16];
      o[dt] = __builtin_amdgcn_wmma_f32_16x16x32_bf16(
          false, pa.v, false, b.v, (short)0, o[dt], false, false);
    }
    __syncthreads();
  }

  // ---- normalize + write context (bf16) ----
  #pragma unroll
  for (int r = 0; r < 8; r++) {
    const float inv  = 1.0f / lrow[r];
    const int   qrow = q0w + r + half * 8;
    unsigned short* orow = ctx + (size_t)qrow * CTX_N + head * DHEAD + l16;
    #pragma unroll
    for (int dt = 0; dt < 16; dt++)
      orow[dt * 16] = f2bf(o[dt][r] * inv);
  }
}

// ---------------------------------------------------------------------------
extern "C" void kernel_launch(void* const* d_in, const int* in_sizes, int n_in,
                              void* d_out, int out_size, void* d_ws, size_t ws_size,
                              hipStream_t stream)
{
  const float* hs = (const float*)d_in[0];
  const float* wq = (const float*)d_in[1];
  const float* wk = (const float*)d_in[2];
  const float* wv = (const float*)d_in[3];
  const float* wo = (const float*)d_in[4];
  const int*  pos = (const int*)d_in[5];

  unsigned short* qkv = (unsigned short*)d_ws;               // 4096 x 8192 bf16
  unsigned short* ctx = qkv + (size_t)S_LEN * QKV_N;         // 4096 x 4096 bf16

  // 1) fused QKV projection (f32 in -> bf16 out)
  {
    dim3 grid(QKV_N / 128, S_LEN / 128);
    gemm_kernel<false, true><<<grid, 256, 0, stream>>>(
        hs, wq, NQH * DHEAD, wk, NKVH * DHEAD, wv, NKVH * DHEAD,
        CTX_N, CTX_N + NKVH * DHEAD, qkv, S_LEN, QKV_N, HDIM);
  }
  // 2) RoPE in place on Q and K
  rope_kernel<<<(S_LEN * 24 * 128) / 256, 256, 0, stream>>>(qkv, pos);
  // 3) sliding-window attention
  {
    dim3 grid(S_LEN / 64, NQH);
    attn_kernel<<<grid, 128, 0, stream>>>(qkv, ctx);
  }
  // 4) output projection (bf16 in -> f32 out)
  {
    dim3 grid(HDIM / 128, S_LEN / 128);
    gemm_kernel<true, false><<<grid, 256, 0, stream>>>(
        ctx, wo, HDIM, wo, HDIM, wo, HDIM,
        HDIM, HDIM, d_out, S_LEN, HDIM, QKV_N);
  }
}
